// HopfieldMemoryLayer_18382460027553
// MI455X (gfx1250) — matrix operand order repaired and checked by reading, not compiled, not run on hardware
//
#include <hip/hip_runtime.h>

// ---------------------------------------------------------------------------
// Types for CDNA5 WMMA (wave32): v_wmma_f32_16x16x32_bf16
// ---------------------------------------------------------------------------
typedef __attribute__((ext_vector_type(16))) __bf16 v16bf;
typedef __attribute__((ext_vector_type(8)))  float  v8f;

union AFrag { v16bf v; unsigned int u[8]; };

__device__ inline unsigned short f2bf(float f) {
  unsigned int u = __float_as_uint(f);
  u += 0x7fffu + ((u >> 16) & 1u);   // round-to-nearest-even
  return (unsigned short)(u >> 16);
}

// ---------------------------------------------------------------------------
// CDNA5 Tensor Data Mover (TDM) support.  Device pass only; falls back to
// cooperative b128 fills when the builtin is unavailable.
// ---------------------------------------------------------------------------
#if defined(__HIP_DEVICE_COMPILE__) && defined(__gfx1250__) && \
    __has_builtin(__builtin_amdgcn_tensor_load_to_lds)
#define HAS_TDM 1
#else
#define HAS_TDM 0
#endif

#if __has_include(<hip/amd_detail/amd_gfx1250_TDM.h>)
#define TDM_ARITY6 1   // therock-10.0 headers -> 6-arg builtin
#else
#define TDM_ARITY6 0   // ROCm 7.2 -> 5-arg builtin
#endif

#if HAS_TDM
typedef __attribute__((ext_vector_type(4))) unsigned int u32x4;
typedef __attribute__((ext_vector_type(8))) int          i32x8;
typedef __attribute__((ext_vector_type(4))) int          i32x4;

// 32-bit LDS byte offset of a pointer into __shared__ space
__device__ inline unsigned ldsOffset(const void* p) {
  return (unsigned)(unsigned long long)
      (__attribute__((address_space(3))) const char*)p;
}

// 2D tile DMA: global (row-major, rowStride elems of bf16) -> LDS, with
// pad_amount dwords of LDS padding appended every pad_interval dwords.
// tileD0 = elements per row, tileD1 = rows. data_size = 2 bytes.
__device__ inline void tdm_load_2d(unsigned ldsAddr, const unsigned short* gptr,
                                   long rowStrideElems, int tensorD0, int tensorD1,
                                   int tileD0, int tileD1,
                                   unsigned padIntervalCode, unsigned padAmountCode) {
  unsigned long long ga = (unsigned long long)gptr;
  u32x4 g0 = { 1u,                                  // count=1, user descriptor
               ldsAddr,                              // D#.lds_addr
               (unsigned)ga,                         // global_addr[31:0]
               ((unsigned)(ga >> 32) & 0x01FFFFFFu)  // global_addr[56:32]
                 | (2u << 30) };                     // type = 2 (image)
  unsigned c0 = (1u << 16)                 // data_size = 1 -> 2 bytes
              | (1u << 20)                 // pad_enable
              | (padIntervalCode << 22)    // dwords before pad = 2^(code+1)
              | (padAmountCode << 25);     // pad dwords = code+1
  unsigned long long s0 = (unsigned long long)rowStrideElems;
  i32x8 g1 = { (int)c0,
               (int)(((unsigned)tensorD0 & 0xFFFFu) << 16),           // td0[15:0]
               (int)((((unsigned)tensorD0 >> 16) & 0xFFFFu)
                     | (((unsigned)tensorD1 & 0xFFFFu) << 16)),       // td0hi|td1lo
               (int)((((unsigned)tensorD1 >> 16) & 0xFFFFu)
                     | ((unsigned)tileD0 << 16)),                     // td1hi|tile0
               (int)(unsigned)tileD1,                                 // tile1|tile2=0
               (int)(unsigned)(s0 & 0xFFFFFFFFu),                     // stride0 lo
               (int)(unsigned)((s0 >> 32) & 0xFFFFu),                 // stride0 hi
               0 };
  i32x4 z4 = {0, 0, 0, 0};
#if TDM_ARITY6
  i32x8 z8 = {0, 0, 0, 0, 0, 0, 0, 0};
  __builtin_amdgcn_tensor_load_to_lds(g0, g1, z4, z4, z8, 0);
#else
  __builtin_amdgcn_tensor_load_to_lds(g0, g1, z4, z4, 0);
#endif
}
#endif  // HAS_TDM

// ---------------------------------------------------------------------------
// f32 -> bf16 cast
// ---------------------------------------------------------------------------
__global__ __launch_bounds__(256)
void cast_f32_bf16_kernel(const float* __restrict__ in,
                          unsigned short* __restrict__ out, long n) {
  long i = (long)blockIdx.x * 256 + threadIdx.x;
  if (i < n) out[i] = f2bf(in[i]);
}

// ---------------------------------------------------------------------------
// RMSNorm one row of E floats -> bf16 (one block per row)
// ---------------------------------------------------------------------------
__global__ __launch_bounds__(256)
void rmsnorm_bf16_kernel(const float* __restrict__ x, const float* __restrict__ w,
                         unsigned short* __restrict__ out, int E) {
  long row = blockIdx.x;
  const float* xr = x + row * E;
  float ss = 0.f;
  for (int e = threadIdx.x; e < E; e += 256) { float v = xr[e]; ss += v * v; }
  for (int m = 16; m >= 1; m >>= 1) ss += __shfl_xor(ss, m, 32);
  __shared__ float red[8];
  if ((threadIdx.x & 31) == 0) red[threadIdx.x >> 5] = ss;
  __syncthreads();
  float tot = 0.f;
  for (int i = 0; i < 8; i++) tot += red[i];
  float scale = rsqrtf(tot / (float)E + 1e-5f);
  for (int e = threadIdx.x; e < E; e += 256)
    out[row * E + e] = f2bf(xr[e] * scale * w[e]);
}

// ---------------------------------------------------------------------------
// C = A(MxK) * B(NxK)^T, bf16 in/out, f32 accumulate via WMMA.
// Block: 256 threads = 8 waves; block tile 128x128.
// TDM path: K-panels (128x32, LDS rows padded to 40 shorts) are DMA'd by the
// Tensor Data Mover, double-buffered, wave 0 issues one step ahead and gates
// the block with s_wait_tensorcnt -> one barrier per K-step, full overlap.
// ---------------------------------------------------------------------------
__global__ __launch_bounds__(256)
void gemm_bf16_nt_kernel(const unsigned short* __restrict__ A,
                         const unsigned short* __restrict__ Bm,
                         unsigned short* __restrict__ C,
                         int Mr, int Nr, int K,
                         long aBatch, long bBatch, long cBatch,
                         int permuteQ, int Sdim, int Hdim, int Ddim) {
  __shared__ __align__(16) unsigned short As[2][128][40];  // 32 + 8 pad
  __shared__ __align__(16) unsigned short Bs[2][128][40];
  int batch = blockIdx.z;
  A  += (long)batch * aBatch;
  Bm += (long)batch * bBatch;
  C  += (long)batch * cBatch;
  int tid  = threadIdx.x;
  int wave = tid >> 5, lane = tid & 31;
  int lane_m = lane & 15, half = lane >> 4;
  int m0 = blockIdx.y * 128;
  int n0 = blockIdx.x * 128;
  int nsteps = K / 32;

  v8f acc[8] = {};

#if HAS_TDM
  if (wave == 0) {
    tdm_load_2d(ldsOffset(&As[0][0][0]), &A[(long)m0 * K], K, K, Mr, 32, 128, 3, 3);
    tdm_load_2d(ldsOffset(&Bs[0][0][0]), &Bm[(long)n0 * K], K, K, Nr, 32, 128, 3, 3);
  }
#endif

  for (int step = 0; step < nsteps; ++step) {
    int cur;
#if HAS_TDM
    cur = step & 1;
    if (wave == 0) __builtin_amdgcn_s_wait_tensorcnt(0);
    __syncthreads();   // publish DMA'd panel; all waves done with prev buffers
    if (wave == 0 && step + 1 < nsteps) {
      int kn = (step + 1) * 32;
      tdm_load_2d(ldsOffset(&As[cur ^ 1][0][0]), &A[(long)m0 * K + kn],
                  K, K, Mr, 32, 128, 3, 3);
      tdm_load_2d(ldsOffset(&Bs[cur ^ 1][0][0]), &Bm[(long)n0 * K + kn],
                  K, K, Nr, 32, 128, 3, 3);
    }
#else
    cur = 0;
    int k0 = step * 32;
    __syncthreads();
    for (int idx = tid; idx < 128 * 4; idx += 256) {   // 128 rows x 4 uint4
      int r = idx >> 2, q = (idx & 3) * 8;
      *(uint4*)&As[0][r][q] = *(const uint4*)&A[(long)(m0 + r) * K + k0 + q];
      *(uint4*)&Bs[0][r][q] = *(const uint4*)&Bm[(long)(n0 + r) * K + k0 + q];
    }
    __syncthreads();
#endif
    // A fragment: 16x32 bf16 (ISA 7.12.2 layout)
    AFrag a;
    #pragma unroll
    for (int v = 0; v < 8; v++) {
      int kk = (v >> 2) * 16 + half * 8 + (v & 3) * 2;
      a.u[v] = *(const unsigned int*)&As[cur][wave * 16 + lane_m][kk];
    }
    #pragma unroll
    for (int nt = 0; nt < 8; nt++) {
      AFrag b;
      #pragma unroll
      for (int v = 0; v < 8; v++) {
        int kk = (v >> 2) * 16 + half * 8 + (v & 3) * 2;
        b.u[v] = *(const unsigned int*)&Bs[cur][nt * 16 + lane_m][kk];
      }
      acc[nt] = __builtin_amdgcn_wmma_f32_16x16x32_bf16(
          false, a.v, false, b.v, (short)0, acc[nt], false, false);
    }
  }

  // C layout: VGPR j <-> row j + 8*half, lane_m = col within 16-tile
  int mrow = m0 + wave * 16;
  #pragma unroll
  for (int nt = 0; nt < 8; nt++) {
    #pragma unroll
    for (int j = 0; j < 8; j++) {
      int row = mrow + j + 8 * half;
      int col = n0 + nt * 16 + lane_m;
      unsigned short bv = f2bf(acc[nt][j]);
      long idx;
      if (permuteQ) {
        int bb = row / Sdim, s = row - bb * Sdim;
        int h  = col / Ddim, d = col - h * Ddim;
        idx = (((long)bb * Hdim + h) * Sdim + s) * (long)Ddim + d;
      } else {
        idx = (long)row * Nr + col;
      }
      C[idx] = bv;
    }
  }
}

// ---------------------------------------------------------------------------
// Fused attention: scores = (Q K^T) * beta/sqrt(D); online softmax over M;
// O = P V.  One block = 4 waves = 64 query rows of one (b,h); K streamed via
// TDM (double-buffered), V transposed through VGPRs; all matmuls WMMA bf16.
// ---------------------------------------------------------------------------
__global__ __launch_bounds__(128)
void hopfield_attn_kernel(const unsigned short* __restrict__ q,     // [B,H,S,D] bf16
                          const unsigned short* __restrict__ keys,  // [H,M,D] bf16
                          const unsigned short* __restrict__ vals,  // [H,M,D] bf16
                          const float* __restrict__ beta,           // [H]
                          float* __restrict__ outr,                 // [B,H,S,D] f32
                          int Bv, int Hv, int Sv, int Mv, int Dv) {
  __shared__ __align__(16) unsigned short Ks[2][64][136];  // K chunk [m][d], +8 pad
  __shared__ __align__(16) unsigned short Vt[128][72];     // V chunk transposed [d][m]
  __shared__ __align__(16) unsigned short Ps[4][16][72];   // per-wave P staging
  int b = blockIdx.z, h = blockIdx.y;
  int tid = threadIdx.x, wave = tid >> 5, lane = tid & 31;
  int lane_m = lane & 15, half = lane >> 4;
  int s0 = blockIdx.x * 64 + wave * 16;
  const unsigned short* qp = q    + (((long)b * Hv + h) * Sv) * Dv;
  const unsigned short* kp = keys + (long)h * Mv * Dv;
  const unsigned short* vp = vals + (long)h * Mv * Dv;
  float bc = fminf(fmaxf(beta[h], 0.01f), 10.f);
  float scale = bc * rsqrtf((float)Dv);

  // Q A-fragments for this wave's 16 rows: 4 k-steps over D=128, kept in VGPRs
  AFrag qa[4];
  #pragma unroll
  for (int kb = 0; kb < 4; kb++) {
    #pragma unroll
    for (int v = 0; v < 8; v++) {
      int d = kb * 32 + (v >> 2) * 16 + half * 8 + (v & 3) * 2;
      qa[kb].u[v] = *(const unsigned int*)&qp[(long)(s0 + lane_m) * Dv + d];
    }
  }

  v8f Oacc[8] = {};
  float rm[8], rl[8];
  #pragma unroll
  for (int j = 0; j < 8; j++) { rm[j] = -1e30f; rl[j] = 0.f; }

#if HAS_TDM
  if (wave == 0)   // prefetch first K chunk: 64 rows x 128 elems, rows padded +4dw
    tdm_load_2d(ldsOffset(&Ks[0][0][0]), kp, Dv, Dv, Mv, 128, 64, 5, 3);
#endif

  int it = 0;
  for (int mc = 0; mc < Mv; mc += 64, ++it) {
    int cur;
#if HAS_TDM
    cur = it & 1;
    if (wave == 0) __builtin_amdgcn_s_wait_tensorcnt(0);
    __syncthreads();
    if (wave == 0 && mc + 64 < Mv)
      tdm_load_2d(ldsOffset(&Ks[cur ^ 1][0][0]), kp + (long)(mc + 64) * Dv,
                  Dv, Dv, Mv, 128, 64, 5, 3);
    // V chunk: transpose through VGPRs (TDM cannot transpose 16-bit data)
    for (int idx = tid; idx < 64 * 16; idx += 128) {
      int r = idx >> 4, dq = (idx & 15) * 8;
      uint4 vv = *(const uint4*)&vp[(long)(mc + r) * Dv + dq];
      Vt[dq + 0][r] = (unsigned short)(vv.x & 0xffffu);
      Vt[dq + 1][r] = (unsigned short)(vv.x >> 16);
      Vt[dq + 2][r] = (unsigned short)(vv.y & 0xffffu);
      Vt[dq + 3][r] = (unsigned short)(vv.y >> 16);
      Vt[dq + 4][r] = (unsigned short)(vv.z & 0xffffu);
      Vt[dq + 5][r] = (unsigned short)(vv.z >> 16);
      Vt[dq + 6][r] = (unsigned short)(vv.w & 0xffffu);
      Vt[dq + 7][r] = (unsigned short)(vv.w >> 16);
    }
    __syncthreads();
#else
    cur = 0;
    __syncthreads();
    for (int idx = tid; idx < 64 * 16; idx += 128) {
      int r = idx >> 4, dq = (idx & 15) * 8;
      *(uint4*)&Ks[0][r][dq] = *(const uint4*)&kp[(long)(mc + r) * Dv + dq];
      uint4 vv = *(const uint4*)&vp[(long)(mc + r) * Dv + dq];
      Vt[dq + 0][r] = (unsigned short)(vv.x & 0xffffu);
      Vt[dq + 1][r] = (unsigned short)(vv.x >> 16);
      Vt[dq + 2][r] = (unsigned short)(vv.y & 0xffffu);
      Vt[dq + 3][r] = (unsigned short)(vv.y >> 16);
      Vt[dq + 4][r] = (unsigned short)(vv.z & 0xffffu);
      Vt[dq + 5][r] = (unsigned short)(vv.z >> 16);
      Vt[dq + 6][r] = (unsigned short)(vv.w & 0xffffu);
      Vt[dq + 7][r] = (unsigned short)(vv.w >> 16);
    }
    __syncthreads();
#endif

    // scores tile: 16 rows x 64 keys, f32 accum
    v8f sc[4] = {};
    #pragma unroll
    for (int kb = 0; kb < 4; kb++) {
      #pragma unroll
      for (int nt = 0; nt < 4; nt++) {
        AFrag bf;
        #pragma unroll
        for (int v = 0; v < 8; v++) {
          int kk = kb * 32 + (v >> 2) * 16 + half * 8 + (v & 3) * 2;
          bf.u[v] = *(const unsigned int*)&Ks[cur][nt * 16 + lane_m][kk];
        }
        sc[nt] = __builtin_amdgcn_wmma_f32_16x16x32_bf16(
            false, qa[kb].v, false, bf.v, (short)0, sc[nt], false, false);
      }
    }

    // scale + per-row chunk max (row j+8*half lives on 16 lanes of one half)
    float cmax[8];
    #pragma unroll
    for (int j = 0; j < 8; j++) cmax[j] = -1e30f;
    #pragma unroll
    for (int nt = 0; nt < 4; nt++)
      #pragma unroll
      for (int j = 0; j < 8; j++) {
        float x = sc[nt][j] * scale;
        sc[nt][j] = x;
        cmax[j] = fmaxf(cmax[j], x);
      }
    #pragma unroll
    for (int j = 0; j < 8; j++)
      for (int m = 1; m < 16; m <<= 1)
        cmax[j] = fmaxf(cmax[j], __shfl_xor(cmax[j], m, 32));

    // online softmax update
    float nmax[8], corr[8], csum[8];
    #pragma unroll
    for (int j = 0; j < 8; j++) {
      nmax[j] = fmaxf(rm[j], cmax[j]);
      corr[j] = __expf(rm[j] - nmax[j]);
      rm[j] = nmax[j];
      csum[j] = 0.f;
    }
    #pragma unroll
    for (int nt = 0; nt < 4; nt++)
      #pragma unroll
      for (int j = 0; j < 8; j++) {
        float p = __expf(sc[nt][j] - nmax[j]);
        csum[j] += p;
        sc[nt][j] = p;
      }
    #pragma unroll
    for (int j = 0; j < 8; j++)
      for (int m = 1; m < 16; m <<= 1) csum[j] += __shfl_xor(csum[j], m, 32);
    #pragma unroll
    for (int j = 0; j < 8; j++) rl[j] = rl[j] * corr[j] + csum[j];
    #pragma unroll
    for (int dt = 0; dt < 8; dt++)
      #pragma unroll
      for (int j = 0; j < 8; j++) Oacc[dt][j] *= corr[j];

    // restage P as bf16 in A-fragment-friendly row-major LDS
    #pragma unroll
    for (int nt = 0; nt < 4; nt++)
      #pragma unroll
      for (int j = 0; j < 8; j++)
        Ps[wave][j + 8 * half][nt * 16 + lane_m] = f2bf(sc[nt][j]);
    __syncthreads();

    // O += P * V  (k over this 64-chunk: 2 steps; 8 d-tiles)
    #pragma unroll
    for (int kb = 0; kb < 2; kb++) {
      AFrag pa;
      #pragma unroll
      for (int v = 0; v < 8; v++) {
        int kk = kb * 32 + (v >> 2) * 16 + half * 8 + (v & 3) * 2;
        pa.u[v] = *(const unsigned int*)&Ps[wave][lane_m][kk];
      }
      #pragma unroll
      for (int dt = 0; dt < 8; dt++) {
        AFrag bv;
        #pragma unroll
        for (int v = 0; v < 8; v++) {
          int kk = kb * 32 + (v >> 2) * 16 + half * 8 + (v & 3) * 2;
          bv.u[v] = *(const unsigned int*)&Vt[dt * 16 + lane_m][kk];
        }
        Oacc[dt] = __builtin_amdgcn_wmma_f32_16x16x32_bf16(
            false, pa.v, false, bv.v, (short)0, Oacc[dt], false, false);
      }
    }
  }

  // finalize: divide by row sums, store retrieved [B,H,S,D] f32
  float* op = outr + (((long)b * Hv + h) * Sv) * Dv;
  #pragma unroll
  for (int dt = 0; dt < 8; dt++) {
    #pragma unroll
    for (int j = 0; j < 8; j++) {
      int srow = s0 + j + 8 * half;
      int d = dt * 16 + lane_m;
      op[(long)srow * Dv + d] = Oacc[dt][j] / rl[j];
    }
  }
}

// ---------------------------------------------------------------------------
// Final: gather retrieved [B,H,S,D] -> RMSNorm over E -> + residual -> out f32
// ---------------------------------------------------------------------------
__global__ __launch_bounds__(256)
void final_out_kernel(const float* __restrict__ retr,
                      const float* __restrict__ qin,
                      const float* __restrict__ w,
                      float* __restrict__ out,
                      int Bv, int Hv, int Sv, int Dv) {
  long row = blockIdx.x;            // b*S + s
  int b = (int)(row / Sv), s = (int)(row - (long)b * Sv);
  const float* base = retr + (((long)b * Hv) * Sv + s) * Dv;
  int E = Hv * Dv;
  long hstride = (long)Sv * Dv;
  float ss = 0.f;
  for (int e = threadIdx.x; e < E; e += 256) {
    int h = e >> 7, d = e & 127;    // D = 128
    float v = base[(long)h * hstride + d];
    ss += v * v;
  }
  for (int m = 16; m >= 1; m >>= 1) ss += __shfl_xor(ss, m, 32);
  __shared__ float red[8];
  if ((threadIdx.x & 31) == 0) red[threadIdx.x >> 5] = ss;
  __syncthreads();
  float tot = 0.f;
  for (int i = 0; i < 8; i++) tot += red[i];
  float scale = rsqrtf(tot / (float)E + 1e-5f);
  for (int e = threadIdx.x; e < E; e += 256) {
    int h = e >> 7, d = e & 127;
    float v = base[(long)h * hstride + d];
    out[row * E + e] = qin[row * E + e] + v * scale * w[e];
  }
}

// ---------------------------------------------------------------------------
// Host launcher
// ---------------------------------------------------------------------------
extern "C" void kernel_launch(void* const* d_in, const int* in_sizes, int n_in,
                              void* d_out, int out_size, void* d_ws, size_t ws_size,
                              hipStream_t stream) {
  const float* qin  = (const float*)d_in[0];  // [B,S,E]
  const float* sp   = (const float*)d_in[1];  // [H,M,D]
  const float* Wq   = (const float*)d_in[2];  // [E,E]
  const float* Wk   = (const float*)d_in[3];  // [D,D]
  const float* Wv   = (const float*)d_in[4];  // [D,D]
  const float* beta = (const float*)d_in[5];  // [1,H,1,1]
  const float* nqw  = (const float*)d_in[6];  // [E]
  const float* nrw  = (const float*)d_in[7];  // [E]
  float* out = (float*)d_out;

  const int B = 4, S = 2048, E = 2048, H = 16, M = 1024, D = 128;

  char* ws = (char*)d_ws;
  auto alloc = [&](size_t bytes) {
    char* p = ws;
    ws += (bytes + 255) & ~(size_t)255;
    return p;
  };
  unsigned short* qn   = (unsigned short*)alloc((size_t)B * S * E * 2);
  unsigned short* wqb  = (unsigned short*)alloc((size_t)E * E * 2);
  unsigned short* spb  = (unsigned short*)alloc((size_t)H * M * D * 2);
  unsigned short* wkb  = (unsigned short*)alloc((size_t)D * D * 2);
  unsigned short* wvb  = (unsigned short*)alloc((size_t)D * D * 2);
  unsigned short* qb   = (unsigned short*)alloc((size_t)B * S * E * 2);
  unsigned short* keyb = (unsigned short*)alloc((size_t)H * M * D * 2);
  unsigned short* valb = (unsigned short*)alloc((size_t)H * M * D * 2);
  float*          retr = (float*)alloc((size_t)B * H * S * D * 4);

  // 1) downconvert weights / patterns to bf16
  {
    long n = (long)E * E;
    cast_f32_bf16_kernel<<<dim3((unsigned)((n + 255) / 256)), dim3(256), 0, stream>>>(Wq, wqb, n);
  }
  {
    long n = (long)H * M * D;
    cast_f32_bf16_kernel<<<dim3((unsigned)((n + 255) / 256)), dim3(256), 0, stream>>>(sp, spb, n);
  }
  {
    long n = (long)D * D;
    cast_f32_bf16_kernel<<<dim3((unsigned)((n + 255) / 256)), dim3(256), 0, stream>>>(Wk, wkb, n);
    cast_f32_bf16_kernel<<<dim3((unsigned)((n + 255) / 256)), dim3(256), 0, stream>>>(Wv, wvb, n);
  }

  // 2) RMSNorm(query_input) -> qn bf16
  rmsnorm_bf16_kernel<<<dim3(B * S), dim3(256), 0, stream>>>(qin, nqw, qn, E);

  // 3) q = qn @ Wq^T  (stored permuted as [B,H,S,D] bf16)
  gemm_bf16_nt_kernel<<<dim3(E / 128, (B * S) / 128, 1), dim3(256), 0, stream>>>(
      qn, wqb, qb, B * S, E, E, 0, 0, 0, /*permuteQ=*/1, S, H, D);

  // 4) keys/values = storedpatterns @ W{k,v}^T per head
  gemm_bf16_nt_kernel<<<dim3(1, M / 128, H), dim3(256), 0, stream>>>(
      spb, wkb, keyb, M, D, D, (long)M * D, 0, (long)M * D, 0, S, H, D);
  gemm_bf16_nt_kernel<<<dim3(1, M / 128, H), dim3(256), 0, stream>>>(
      spb, wvb, valb, M, D, D, (long)M * D, 0, (long)M * D, 0, S, H, D);

  // 5) fused attention (flash-style over M) -> retrieved [B,H,S,D] f32
  hopfield_attn_kernel<<<dim3(S / 64, H, B), dim3(128), 0, stream>>>(
      qb, keyb, valb, beta, retr, B, H, S, M, D);

  // 6) RMSNorm(retrieved) + residual add -> out f32
  final_out_kernel<<<dim3(B * S), dim3(256), 0, stream>>>(retr, qin, nrw, out, B, H, S, D);
}
